// GAT_GCN_23029614641254
// MI455X (gfx1250) — matrix operand order repaired and checked by reading, not compiled, not run on hardware
//
#include <hip/hip_runtime.h>

// ---------------------------------------------------------------------------
// Types for CDNA5 WMMA (wave32): v_wmma_f32_16x16x32_bf16
// ---------------------------------------------------------------------------
typedef __bf16 bf16;
typedef __attribute__((ext_vector_type(16))) __bf16 v16bf;
typedef __attribute__((ext_vector_type(8)))  __bf16 v8bf;
typedef __attribute__((ext_vector_type(8)))  float  v8f;
typedef __attribute__((__vector_size__(4 * sizeof(int)))) int v4i;

#define AS1 __attribute__((address_space(1)))
#define AS3 __attribute__((address_space(3)))

#define DEV static __device__ __forceinline__

#if defined(__has_builtin)
#if __has_builtin(__builtin_amdgcn_global_load_async_to_lds_b128)
#define HAVE_ASYNC_LDS 1
#endif
#endif

// round-to-nearest-even f32 -> bf16 (stored as ushort; only WMMA reads it back)
DEV unsigned short f2bf(float f) {
  unsigned int u = __float_as_uint(f);
  u += 0x7FFFu + ((u >> 16) & 1u);
  return (unsigned short)(u >> 16);
}

// float atomic max via int/uint ordering trick (works for all finite values)
DEV void atomicMaxF(float* addr, float v) {
  if (v >= 0.0f) atomicMax((int*)addr, __float_as_int(v));
  else           atomicMin((unsigned int*)addr, __float_as_uint(v));
}

// ---------------------------------------------------------------------------
// Problem constants
// ---------------------------------------------------------------------------
#define NN   20000
#define MN   20096        /* NN padded to multiple of 128 */
#define EE   160000
#define ETOT 180000       /* E + self loops */
#define HH   10
#define FF   78
#define FP   96           /* 78 -> mult of 32 (K dim) */
#define HF   780
#define HFP  832          /* 780 -> mult of 64 (N dim) */
#define HFK  800          /* 780 -> mult of 32 (K dim) */
#define BB   128
#define XG_K 1568         /* 1560 -> mult of 32 */
#define G1N  1536         /* 1500 -> mult of 64 (N) and 32 (K) */
#define LL   1000
#define LP   1024
#define EDIM 128
#define CONVO 121
#define NFLT 32
#define CFK  3872         /* 32*121, already mult of 32 */

// ---------------------------------------------------------------------------
// LDS-staged bf16 WMMA GEMM:  C[M,N] = A[M,K] @ Bt[N,K]^T (+bias, relu)
// block = 256 threads = 8 waves; block tile 128x64.
// Each wave owns 16 rows x 64 cols = 4 accumulators; one A fragment is
// reused across 4 WMMAs (reuse_a hint set on the identical repeats).
// Staging uses GLOBAL_LOAD_ASYNC_TO_LDS_B128 (ASYNCcnt) when available.
// M % 128 == 0, N % 64 == 0, K % 32 == 0 (caller pads buffers).
// ---------------------------------------------------------------------------
__global__ __launch_bounds__(256) void k_gemm_bf16(
    const unsigned short* __restrict__ A, const unsigned short* __restrict__ Bt,
    float* __restrict__ C, int lda, int ldb, int ldc, int K,
    const float* __restrict__ bias, int nbias, int relu,
    long long sA, long long sB, long long sC)
{
  __shared__ __align__(32) unsigned short As[128 * 32];  // 8 KB
  __shared__ __align__(32) unsigned short Bs[64 * 32];   // 4 KB

  const unsigned short* Ab = A  + sA * blockIdx.z;
  const unsigned short* Bb = Bt + sB * blockIdx.z;
  float*                Cb = C  + sC * blockIdx.z;

  const int tid  = threadIdx.x;
  const int wave = tid >> 5;
  const int lane = tid & 31;
  const int m0   = blockIdx.x * 128;
  const int n0   = blockIdx.y * 64;
  const int ar   = tid >> 1;     // A staging: row 0..127
  const int ah   = tid & 1;      // A staging: 16-elem half
  const int br   = tid >> 2;     // B staging: row 0..63
  const int bs   = tid & 3;      // B staging: 8-elem segment
  const int row  = lane & 15;    // fragment row within wave tile
  const int hi   = lane >> 4;    // lane group (selects contiguous K half 16*hi..16*hi+15)

  const long long aoff = (long long)(m0 + ar) * lda + ah * 16;
  const long long boff = (long long)(n0 + br) * ldb + bs * 8;

  v8f acc0 = {0.f,0.f,0.f,0.f,0.f,0.f,0.f,0.f};
  v8f acc1 = acc0, acc2 = acc0, acc3 = acc0;

  for (int k0 = 0; k0 < K; k0 += 32) {
    // ---- stage A 128x32 + B 64x32 bf16 tiles (16B per thread each) ----
#if HAVE_ASYNC_LDS
    __builtin_amdgcn_global_load_async_to_lds_b128(
        (AS1 v4i*)&Ab[aoff + k0],
        (AS3 v4i*)&As[ar * 32 + ah * 16], 0, 0);
    __builtin_amdgcn_global_load_async_to_lds_b128(
        (AS1 v4i*)&Bb[boff + k0],
        (AS3 v4i*)&Bs[br * 32 + bs * 8], 0, 0);
#else
    *(uint4*)&As[ar * 32 + ah * 16] = *(const uint4*)&Ab[aoff + k0];
    *(uint4*)&Bs[br * 32 + bs * 8]  = *(const uint4*)&Bb[boff + k0];
#endif
    if (k0 + 32 < K) {  // prefetch next tiles -> global_prefetch_b8
      __builtin_prefetch(&Ab[aoff + k0 + 32], 0, 1);
      __builtin_prefetch(&Bb[boff + k0 + 32], 0, 1);
    }
#if HAVE_ASYNC_LDS
    asm volatile("s_wait_asynccnt 0x0" ::: "memory");
#endif
    __syncthreads();

    // ---- fragments ----
    // A: lane<16 -> row, K=0..15; lane>=16 -> row, K=16..31
    const bf16* ap = (const bf16*)&As[(wave * 16 + row) * 32];
    v8bf a0 = *(const v8bf*)(ap + hi * 16);
    v8bf a1 = *(const v8bf*)(ap + hi * 16 + 8);
    v16bf a = __builtin_shufflevector(a0, a1, 0,1,2,3,4,5,6,7,8,9,10,11,12,13,14,15);
    // B subtile j: lane n (0..15) holds col j*16+n, K = hi*16 .. hi*16+15
    const bf16* bp = (const bf16*)Bs;
    v16bf b0 = *(const v16bf*)(bp + ((0 * 16 + row) * 32) + hi * 16);
    v16bf b1 = *(const v16bf*)(bp + ((1 * 16 + row) * 32) + hi * 16);
    v16bf b2 = *(const v16bf*)(bp + ((2 * 16 + row) * 32) + hi * 16);
    v16bf b3 = *(const v16bf*)(bp + ((3 * 16 + row) * 32) + hi * 16);

    // 4 WMMAs sharing A; reuse_a hint on the identical repeats
    acc0 = __builtin_amdgcn_wmma_f32_16x16x32_bf16(false, a, false, b0, (short)0, acc0, false, false);
    acc1 = __builtin_amdgcn_wmma_f32_16x16x32_bf16(false, a, false, b1, (short)0, acc1, true,  false);
    acc2 = __builtin_amdgcn_wmma_f32_16x16x32_bf16(false, a, false, b2, (short)0, acc2, true,  false);
    acc3 = __builtin_amdgcn_wmma_f32_16x16x32_bf16(false, a, false, b3, (short)0, acc3, true,  false);
    __syncthreads();
  }

  // ---- epilogue: C layout per tile: VGPR v -> M = v + 8*hi, N = lane&15 ----
  const int mb = m0 + wave * 16 + hi * 8;
  v8f accs[4] = {acc0, acc1, acc2, acc3};
  for (int j = 0; j < 4; ++j) {
    const int n = n0 + j * 16 + (lane & 15);
    const float bv = (bias && n < nbias) ? bias[n] : 0.f;
    for (int v = 0; v < 8; ++v) {
      float val = accs[j][v] + bv;
      if (relu) val = fmaxf(val, 0.f);
      Cb[(long long)(mb + v) * ldc + n] = val;
    }
  }
}

// ---------------------------------------------------------------------------
// Elementwise / conversion / scatter kernels
// ---------------------------------------------------------------------------
__global__ void k_fill(float* p, float v, long long n) {
  long long i = (long long)blockIdx.x * 256 + threadIdx.x;
  if (i < n) p[i] = v;
}

// dst[r,c] (Md x Kd, row-major) = src[r,c] if in-range else 0, as bf16
__global__ void k_cvt_pad(const float* __restrict__ src, unsigned short* __restrict__ dst,
                          int Ms, int Ks, int lds, int Md, int Kd) {
  long long i = (long long)blockIdx.x * 256 + threadIdx.x;
  if (i >= (long long)Md * Kd) return;
  int rr = (int)(i / Kd), c = (int)(i % Kd);
  float v = (rr < Ms && c < Ks) ? src[(long long)rr * lds + c] : 0.f;
  dst[i] = f2bf(v);
}

// transpose-convert weight W[Ksrc,Nsrc] -> Bt[n,k] (Nd x Kd) bf16, zero pad
__global__ void k_cvt_t(const float* __restrict__ src, unsigned short* __restrict__ dst,
                        int Ksrc, int Nsrc, int Nd, int Kd) {
  long long i = (long long)blockIdx.x * 256 + threadIdx.x;
  if (i >= (long long)Nd * Kd) return;
  int n = (int)(i / Kd), k = (int)(i % Kd);
  float v = (k < Ksrc && n < Nsrc) ? src[(long long)k * Nsrc + n] : 0.f;
  dst[i] = f2bf(v);
}

// Wc[32,1000,8] -> Wstack[(k*32+o), i] (256 x 1024) bf16
__global__ void k_wstack(const float* __restrict__ Wc, unsigned short* __restrict__ dst) {
  int i = blockIdx.x * 256 + threadIdx.x;
  if (i >= 256 * 1024) return;
  int m = i >> 10, c = i & 1023;
  int k = m >> 5, o = m & 31;
  float v = (c < LL) ? Wc[o * (LL * 8) + c * 8 + k] : 0.f;
  dst[i] = f2bf(v);
}

// et_t[b, d, i] = emb[target[b,i], d]  (B x 128 x 1024) bf16, zero pad i>=1000
__global__ void k_embed(const int* __restrict__ target, const float* __restrict__ emb,
                        unsigned short* __restrict__ dst) {
  long long i = (long long)blockIdx.x * 256 + threadIdx.x;
  if (i >= (long long)BB * EDIM * LP) return;
  int ii = (int)(i & (LP - 1));
  int d  = (int)((i >> 10) & 127);
  int b  = (int)(i >> 17);
  float v = 0.f;
  if (ii < LL) { int tok = target[b * LL + ii]; v = emb[tok * EDIM + d]; }
  dst[i] = f2bf(v);
}

// attention logits per node/head
__global__ void k_att(const float* __restrict__ h, const float* __restrict__ att_src,
                      const float* __restrict__ att_dst, float* a_s, float* a_d) {
  int i = blockIdx.x * 256 + threadIdx.x;
  if (i >= NN * HH) return;
  int n = i / HH, hd = i % HH;
  const float* hp = h + (long long)n * HFP + hd * FF;
  const float* ws = att_src + hd * FF;
  const float* wd = att_dst + hd * FF;
  float s1 = 0.f, s2 = 0.f;
  for (int f = 0; f < FF; ++f) { float hv = hp[f]; s1 += hv * ws[f]; s2 += hv * wd[f]; }
  a_s[i] = s1; a_d[i] = s2;
}

DEV void edge_ends(const int* ei, int e, int& s, int& d) {
  if (e < EE) { s = ei[e]; d = ei[EE + e]; }
  else        { s = e - EE; d = s; }            // self loops
}

DEV float edge_logit(const float* a_s, const float* a_d, int s, int d, int hd) {
  float v = a_s[s * HH + hd] + a_d[d * HH + hd];
  return v > 0.f ? v : 0.2f * v;                // leaky_relu(0.2)
}

__global__ void k_edge_max(const int* ei, const float* a_s, const float* a_d, float* m) {
  int i = blockIdx.x * 256 + threadIdx.x;
  if (i >= ETOT * HH) return;
  int e = i / HH, hd = i % HH, s, d;
  edge_ends(ei, e, s, d);
  atomicMaxF(&m[d * HH + hd], edge_logit(a_s, a_d, s, d, hd));
}

__global__ void k_edge_expsum(const int* ei, const float* a_s, const float* a_d,
                              const float* m, float* ssum) {
  int i = blockIdx.x * 256 + threadIdx.x;
  if (i >= ETOT * HH) return;
  int e = i / HH, hd = i % HH, s, d;
  edge_ends(ei, e, s, d);
  float v = edge_logit(a_s, a_d, s, d, hd);
  atomicAdd(&ssum[d * HH + hd], __expf(v - m[d * HH + hd]));
}

__global__ void k_edge_msg(const int* ei, const float* a_s, const float* a_d,
                           const float* m, const float* ssum,
                           const float* __restrict__ h, float* x1) {
  int i = blockIdx.x * 256 + threadIdx.x;
  if (i >= ETOT * HH) return;
  int e = i / HH, hd = i % HH, s, d;
  edge_ends(ei, e, s, d);
  float v = edge_logit(a_s, a_d, s, d, hd);
  float alpha = __expf(v - m[d * HH + hd]) / (ssum[d * HH + hd] + 1e-16f);
  const float* hs = h + (long long)s * HFP + hd * FF;
  float*       xd = x1 + (long long)d * HFP + hd * FF;
  for (int f = 0; f < FF; ++f) atomicAdd(&xd[f], hs[f] * alpha);
}

__global__ void k_bias_relu(float* p, const float* bias, int M, int Nc, int ld) {
  long long i = (long long)blockIdx.x * 256 + threadIdx.x;
  if (i >= (long long)M * Nc) return;
  int rr = (int)(i / Nc), c = (int)(i % Nc);
  float v = p[(long long)rr * ld + c] + bias[c];
  p[(long long)rr * ld + c] = fmaxf(v, 0.f);
}

__global__ void k_deg(const int* ei, float* deg) {
  int e = blockIdx.x * 256 + threadIdx.x;
  if (e >= ETOT) return;
  int s, d; edge_ends(ei, e, s, d);
  atomicAdd(&deg[d], 1.0f);
}

__global__ void k_gcn(const int* ei, const float* deg, const float* __restrict__ xw,
                      float* x2) {
  long long i = (long long)blockIdx.x * 256 + threadIdx.x;
  if (i >= (long long)ETOT * HF) return;
  int e = (int)(i / HF), f = (int)(i % HF);
  int s, d; edge_ends(ei, e, s, d);
  float norm = rsqrtf(deg[s]) * rsqrtf(deg[d]);
  atomicAdd(&x2[(long long)d * HFP + f], xw[(long long)s * HFP + f] * norm);
}

__global__ void k_pool(const int* batch, const float* __restrict__ x2,
                       float* gmx, float* gap) {
  long long i = (long long)blockIdx.x * 256 + threadIdx.x;
  if (i >= (long long)NN * HF) return;
  int n = (int)(i / HF), f = (int)(i % HF);
  int g = batch[n];
  float v = x2[(long long)n * HFP + f];
  atomicMaxF(&gmx[g * HF + f], v);
  atomicAdd(&gap[g * HF + f], v);
}

__global__ void k_cnt(const int* batch, float* cnt) {
  int n = blockIdx.x * 256 + threadIdx.x;
  if (n >= NN) return;
  atomicAdd(&cnt[batch[n]], 1.0f);
}

// xg_bf[g, c] : c<780 max-pool, 780..1559 mean-pool, pad to 1568
__global__ void k_pool_fin(const float* gmx, const float* gap, const float* cnt,
                           unsigned short* xg) {
  int i = blockIdx.x * 256 + threadIdx.x;
  if (i >= BB * XG_K) return;
  int g = i / XG_K, c = i % XG_K;
  float v = 0.f;
  if (c < HF)            { v = gmx[g * HF + c]; if (!(v > -1e29f)) v = 0.f; }
  else if (c < 2 * HF)   { v = gap[g * HF + (c - HF)] / fmaxf(cnt[g], 1.f); }
  xg[i] = f2bf(v);
}

// conv[b,o,t] = bc[o] + sum_k Z[b, k*32+o, t+k]; flatten (o*121+t) as bf16
__global__ void k_shiftsum(const float* __restrict__ Z, const float* bc,
                           unsigned short* cf) {
  int i = blockIdx.x * 256 + threadIdx.x;
  if (i >= BB * NFLT * CONVO) return;
  int t = i % CONVO;
  int o = (i / CONVO) % NFLT;
  int b = i / (CONVO * NFLT);
  float v = bc[o];
  for (int k = 0; k < 8; ++k)
    v += Z[(long long)b * (256 * 128) + (k * 32 + o) * 128 + (t + k)];
  cf[(long long)b * CFK + o * CONVO + t] = f2bf(v);
}

__global__ void k_concat(const float* xg2, const float* xt, unsigned short* xc) {
  int i = blockIdx.x * 256 + threadIdx.x;
  if (i >= BB * 256) return;
  int b = i / 256, c = i % 256;
  float v = (c < 128) ? xg2[b * 128 + c] : xt[b * 128 + (c - 128)];
  xc[i] = f2bf(v);
}

__global__ void k_out(const float* outc, const float* bo, float* out) {
  int b = blockIdx.x * 256 + threadIdx.x;
  if (b >= BB) return;
  out[b] = outc[b * 64] + bo[0];
}

// ---------------------------------------------------------------------------
// Host launcher
// ---------------------------------------------------------------------------
static inline unsigned gb(long long n) { return (unsigned)((n + 255) / 256); }

extern "C" void kernel_launch(void* const* d_in, const int* in_sizes, int n_in,
                              void* d_out, int out_size, void* d_ws, size_t ws_size,
                              hipStream_t stream) {
  (void)in_sizes; (void)n_in; (void)out_size; (void)ws_size;

  const float* x       = (const float*)d_in[0];
  const int*   ei      = (const int*)  d_in[1];
  const int*   batch   = (const int*)  d_in[2];
  const int*   target  = (const int*)  d_in[3];
  const float* W1      = (const float*)d_in[4];
  const float* att_src = (const float*)d_in[5];
  const float* att_dst = (const float*)d_in[6];
  const float* b1      = (const float*)d_in[7];
  const float* W2      = (const float*)d_in[8];
  const float* b2      = (const float*)d_in[9];
  const float* Wg1     = (const float*)d_in[10];
  const float* bg1     = (const float*)d_in[11];
  const float* Wg2     = (const float*)d_in[12];
  const float* bg2     = (const float*)d_in[13];
  const float* emb     = (const float*)d_in[14];
  const float* Wc      = (const float*)d_in[15];
  const float* bc      = (const float*)d_in[16];
  const float* Wxt     = (const float*)d_in[17];
  const float* bxt     = (const float*)d_in[18];
  const float* Wf1     = (const float*)d_in[19];
  const float* bf1     = (const float*)d_in[20];
  const float* Wf2     = (const float*)d_in[21];
  const float* bf2     = (const float*)d_in[22];
  const float* Wo      = (const float*)d_in[23];
  const float* bo      = (const float*)d_in[24];
  float* out = (float*)d_out;

  char* ws = (char*)d_ws; size_t off = 0;
  auto alloc = [&](size_t bytes) -> void* {
    void* p = ws + off; off = (off + bytes + 255) & ~(size_t)255; return p;
  };
  typedef unsigned short us;

  us*    x_bf    = (us*)   alloc((size_t)MN * FP * 2);
  us*    W1t     = (us*)   alloc((size_t)HFP * FP * 2);
  float* h       = (float*)alloc((size_t)MN * HFP * 4);   // reused for xw
  float* a_s     = (float*)alloc((size_t)NN * HH * 4);
  float* a_d     = (float*)alloc((size_t)NN * HH * 4);
  float* mx      = (float*)alloc((size_t)NN * HH * 4);
  float* ssum    = (float*)alloc((size_t)NN * HH * 4);
  float* x1      = (float*)alloc((size_t)MN * HFP * 4);   // reused for x2
  us*    x1_bf   = (us*)   alloc((size_t)MN * HFK * 2);
  us*    W2t     = (us*)   alloc((size_t)HFP * HFK * 2);
  float* deg     = (float*)alloc((size_t)NN * 4);
  float* gmx     = (float*)alloc((size_t)BB * HF * 4);
  float* gap     = (float*)alloc((size_t)BB * HF * 4);
  float* cnt     = (float*)alloc((size_t)BB * 4);
  us*    xg_bf   = (us*)   alloc((size_t)BB * XG_K * 2);
  us*    Wg1t    = (us*)   alloc((size_t)G1N * XG_K * 2);
  float* xg1     = (float*)alloc((size_t)BB * G1N * 4);
  us*    xg1_bf  = (us*)   alloc((size_t)BB * G1N * 2);
  us*    Wg2t    = (us*)   alloc((size_t)128 * G1N * 2);
  float* xg2     = (float*)alloc((size_t)BB * 128 * 4);
  us*    et_bf   = (us*)   alloc((size_t)BB * EDIM * LP * 2);
  us*    Wstk    = (us*)   alloc((size_t)256 * LP * 2);
  float* Z       = (float*)alloc((size_t)BB * 256 * 128 * 4);
  us*    cf_bf   = (us*)   alloc((size_t)BB * CFK * 2);
  us*    Wxtt    = (us*)   alloc((size_t)128 * CFK * 2);
  float* xt      = (float*)alloc((size_t)BB * 128 * 4);
  us*    xc_bf   = (us*)   alloc((size_t)BB * 256 * 2);
  us*    Wf1t    = (us*)   alloc((size_t)1024 * 256 * 2);
  float* xf1     = (float*)alloc((size_t)BB * 1024 * 4);
  us*    xf1_bf  = (us*)   alloc((size_t)BB * 1024 * 2);
  us*    Wf2t    = (us*)   alloc((size_t)512 * 1024 * 2);
  float* xf2     = (float*)alloc((size_t)BB * 512 * 4);
  us*    xf2_bf  = (us*)   alloc((size_t)BB * 512 * 2);
  us*    Wot     = (us*)   alloc((size_t)64 * 512 * 2);
  float* outc    = (float*)alloc((size_t)BB * 64 * 4);
  float* xw      = h;   // reuse
  float* x2      = x1;  // reuse

  // ---- GAT: h = x @ W1 -------------------------------------------------
  k_cvt_pad<<<gb((long long)MN * FP), 256, 0, stream>>>(x, x_bf, NN, FF, FF, MN, FP);
  k_cvt_t  <<<gb((long long)HFP * FP), 256, 0, stream>>>(W1, W1t, FF, HF, HFP, FP);
  k_gemm_bf16<<<dim3(MN / 128, HFP / 64, 1), 256, 0, stream>>>(
      x_bf, W1t, h, FP, FP, HFP, FP, nullptr, 0, 0, 0, 0, 0);

  k_att<<<gb(NN * HH), 256, 0, stream>>>(h, att_src, att_dst, a_s, a_d);

  k_fill<<<gb(NN * HH), 256, 0, stream>>>(mx, -1e30f, NN * HH);
  k_fill<<<gb(NN * HH), 256, 0, stream>>>(ssum, 0.f, NN * HH);
  k_fill<<<gb((long long)MN * HFP), 256, 0, stream>>>(x1, 0.f, (long long)MN * HFP);
  k_fill<<<gb(NN), 256, 0, stream>>>(deg, 0.f, NN);

  k_edge_max   <<<gb(ETOT * HH), 256, 0, stream>>>(ei, a_s, a_d, mx);
  k_edge_expsum<<<gb(ETOT * HH), 256, 0, stream>>>(ei, a_s, a_d, mx, ssum);
  k_edge_msg   <<<gb(ETOT * HH), 256, 0, stream>>>(ei, a_s, a_d, mx, ssum, h, x1);
  k_bias_relu  <<<gb((long long)NN * HF), 256, 0, stream>>>(x1, b1, NN, HF, HFP);

  // ---- GCN: xw = x1 @ W2, symmetric-norm scatter -----------------------
  k_cvt_pad<<<gb((long long)MN * HFK), 256, 0, stream>>>(x1, x1_bf, NN, HF, HFP, MN, HFK);
  k_cvt_t  <<<gb((long long)HFP * HFK), 256, 0, stream>>>(W2, W2t, HF, HF, HFP, HFK);
  k_gemm_bf16<<<dim3(MN / 128, HFP / 64, 1), 256, 0, stream>>>(
      x1_bf, W2t, xw, HFK, HFK, HFP, HFK, nullptr, 0, 0, 0, 0, 0);

  k_deg<<<gb(ETOT), 256, 0, stream>>>(ei, deg);
  k_fill<<<gb((long long)MN * HFP), 256, 0, stream>>>(x2, 0.f, (long long)MN * HFP);
  k_gcn<<<gb((long long)ETOT * HF), 256, 0, stream>>>(ei, deg, xw, x2);
  k_bias_relu<<<gb((long long)NN * HF), 256, 0, stream>>>(x2, b2, NN, HF, HFP);

  // ---- pooling + graph MLP --------------------------------------------
  k_fill<<<gb(BB * HF), 256, 0, stream>>>(gmx, -1e30f, BB * HF);
  k_fill<<<gb(BB * HF), 256, 0, stream>>>(gap, 0.f, BB * HF);
  k_fill<<<gb(BB), 256, 0, stream>>>(cnt, 0.f, BB);
  k_pool<<<gb((long long)NN * HF), 256, 0, stream>>>(batch, x2, gmx, gap);
  k_cnt <<<gb(NN), 256, 0, stream>>>(batch, cnt);
  k_pool_fin<<<gb(BB * XG_K), 256, 0, stream>>>(gmx, gap, cnt, xg_bf);

  k_cvt_t<<<gb((long long)G1N * XG_K), 256, 0, stream>>>(Wg1, Wg1t, 2 * HF, 1500, G1N, XG_K);
  k_gemm_bf16<<<dim3(1, G1N / 64, 1), 256, 0, stream>>>(
      xg_bf, Wg1t, xg1, XG_K, XG_K, G1N, XG_K, bg1, 1500, 1, 0, 0, 0);
  k_cvt_pad<<<gb((long long)BB * G1N), 256, 0, stream>>>(xg1, xg1_bf, BB, 1500, G1N, BB, G1N);
  k_cvt_t<<<gb((long long)128 * G1N), 256, 0, stream>>>(Wg2, Wg2t, 1500, 128, 128, G1N);
  k_gemm_bf16<<<dim3(1, 128 / 64, 1), 256, 0, stream>>>(
      xg1_bf, Wg2t, xg2, G1N, G1N, 128, G1N, bg2, 128, 0, 0, 0, 0);

  // ---- protein branch: embed -> batched WMMA conv -> shift-sum ---------
  k_embed<<<gb((long long)BB * EDIM * LP), 256, 0, stream>>>(target, emb, et_bf);
  k_wstack<<<gb(256 * 1024), 256, 0, stream>>>(Wc, Wstk);
  k_gemm_bf16<<<dim3(256 / 128, 128 / 64, BB), 256, 0, stream>>>(
      Wstk, et_bf, Z, LP, LP, 128, LP, nullptr, 0, 0,
      0, (long long)EDIM * LP, (long long)256 * 128);
  k_shiftsum<<<gb(BB * NFLT * CONVO), 256, 0, stream>>>(Z, bc, cf_bf);

  k_cvt_t<<<gb((long long)128 * CFK), 256, 0, stream>>>(Wxt, Wxtt, CFK, 128, 128, CFK);
  k_gemm_bf16<<<dim3(1, 128 / 64, 1), 256, 0, stream>>>(
      cf_bf, Wxtt, xt, CFK, CFK, 128, CFK, bxt, 128, 0, 0, 0, 0);

  // ---- joint head ------------------------------------------------------
  k_concat<<<gb(BB * 256), 256, 0, stream>>>(xg2, xt, xc_bf);
  k_cvt_t<<<gb((long long)1024 * 256), 256, 0, stream>>>(Wf1, Wf1t, 256, 1024, 1024, 256);
  k_gemm_bf16<<<dim3(1, 1024 / 64, 1), 256, 0, stream>>>(
      xc_bf, Wf1t, xf1, 256, 256, 1024, 256, bf1, 1024, 1, 0, 0, 0);
  k_cvt_pad<<<gb((long long)BB * 1024), 256, 0, stream>>>(xf1, xf1_bf, BB, 1024, 1024, BB, 1024);
  k_cvt_t<<<gb((long long)512 * 1024), 256, 0, stream>>>(Wf2, Wf2t, 1024, 512, 512, 1024);
  k_gemm_bf16<<<dim3(1, 512 / 64, 1), 256, 0, stream>>>(
      xf1_bf, Wf2t, xf2, 1024, 1024, 512, 1024, bf2, 512, 1, 0, 0, 0);
  k_cvt_pad<<<gb((long long)BB * 512), 256, 0, stream>>>(xf2, xf2_bf, BB, 512, 512, BB, 512);
  k_cvt_t<<<gb((long long)64 * 512), 256, 0, stream>>>(Wo, Wot, 512, 1, 64, 512);
  k_gemm_bf16<<<dim3(1, 1, 1), 256, 0, stream>>>(
      xf2_bf, Wot, outc, 512, 512, 64, 512, nullptr, 0, 0, 0, 0, 0);
  k_out<<<gb(BB), 256, 0, stream>>>(outc, bo, out);
}
